// BertLayer_30605936951582
// MI455X (gfx1250) — matrix operand order repaired and checked by reading, compiled, not run on hardware
//
#include <hip/hip_runtime.h>
#include <hip/hip_bf16.h>
#include <math.h>

// ---- problem dims ----
#define B_     64
#define S_     512
#define HID_   40
#define HEADS_ 4
#define DH_    10
#define INTER_ 20
#define DHP    16   // padded head dim
#define KP     64   // padded K for HID-sized GEMMs
#define EPS_   1e-5f

typedef __bf16 v16bf __attribute__((ext_vector_type(16)));
typedef __bf16 v8bf  __attribute__((ext_vector_type(8)));
typedef float  v8f   __attribute__((ext_vector_type(8)));
typedef unsigned int u32x4 __attribute__((ext_vector_type(4)));
typedef int          i32x4 __attribute__((ext_vector_type(4)));
typedef int          i32x8 __attribute__((ext_vector_type(8)));

__device__ __forceinline__ __bf16 f2bf(float f) {
  unsigned u = __builtin_bit_cast(unsigned, f);
  unsigned r = (u + 0x7FFFu + ((u >> 16) & 1u)) >> 16;
  unsigned short h = (unsigned short)r;
  return __builtin_bit_cast(__bf16, h);
}

__device__ __forceinline__ v8f wmma_bf16(v16bf a, v16bf b, v8f c) {
  return __builtin_amdgcn_wmma_f32_16x16x32_bf16(false, a, false, b, (short)0, c,
                                                 false, false);
}

__device__ __forceinline__ v16bf zero_v16bf() {
  v16bf z;
  __bf16 zb = f2bf(0.0f);
#pragma unroll
  for (int i = 0; i < 16; ++i) z[i] = zb;
  return z;
}

// Issue a 2D TDM tensor_load_to_lds (unconditional: compile success == tensor op
// emitted). Descriptor per CDNA5 ISA ch8:
// group0: [1:0]=count=1, [63:32]=lds_addr, [120:64]=global_addr, [127:126]=type=2
// group1: [17:16]=data_size code, tensor_dim0/1, tile_dim0/1, tensor_dim0_stride
__device__ __forceinline__ void tdm_load_2d(unsigned lds_off, const void* gptr,
                                            unsigned ds_code,  // 0:1B 1:2B 2:4B
                                            unsigned dim0, unsigned dim1,
                                            unsigned stride0) {
  unsigned long long ga = (unsigned long long)(size_t)gptr;
  u32x4 g0;
  g0[0] = 1u;                                    // count=1, user mode
  g0[1] = lds_off;                               // lds_addr (bytes)
  g0[2] = (unsigned)(ga & 0xFFFFFFFFu);          // global_addr[31:0]
  g0[3] = (unsigned)((ga >> 32) & 0x01FFFFFFu) | 0x80000000u;  // addr[56:32]|type=2
  i32x8 g1;
  g1[0] = (int)(ds_code << 16);                  // workgroup_mask=0, data_size
  g1[1] = (int)(dim0 << 16);                     // tensor_dim0[15:0]
  g1[2] = (int)((dim0 >> 16) | (dim1 << 16));    // tensor_dim0[31:16]|tensor_dim1[15:0]
  g1[3] = (int)((dim1 >> 16) | (dim0 << 16));    // tensor_dim1[31:16]|tile_dim0
  g1[4] = (int)dim1;                             // tile_dim1 (tile_dim2=0)
  g1[5] = (int)stride0;                          // tensor_dim0_stride[31:0]
  g1[6] = 0;
  g1[7] = 0;
  i32x4 gz;
  gz[0] = 0; gz[1] = 0; gz[2] = 0; gz[3] = 0;
#if __clang_major__ >= 23
  i32x8 gz8;
  gz8[0]=0; gz8[1]=0; gz8[2]=0; gz8[3]=0; gz8[4]=0; gz8[5]=0; gz8[6]=0; gz8[7]=0;
  __builtin_amdgcn_tensor_load_to_lds(g0, g1, gz, gz, gz8, 0);
#else
  __builtin_amdgcn_tensor_load_to_lds(g0, g1, gz, gz, 0);
#endif
}

// A fragment (16x32 bf16) from row-major bf16 buffer [16][ldk] (16B-aligned rows),
// K window ks*32..+31. Lanes 0-15 (M=lane) hold K 0..15; lanes 16-31
// hold K 16..31. Two ds_load_b128 + register concat.
__device__ __forceinline__ v16bf load_a(const __bf16* Abuf, int ldk, int ks, int lane) {
  int m = lane & 15;
  const __bf16* row = Abuf + m * ldk + ks * 32 + ((lane < 16) ? 0 : 16);
  v8bf lo = *(const v8bf*)row;
  v8bf hi = *(const v8bf*)(row + 8);
  return __builtin_shufflevector(lo, hi, 0, 1, 2, 3, 4, 5, 6, 7, 8, 9, 10, 11,
                                 12, 13, 14, 15);
}

// B fragment (32x16 bf16) from pre-packed transposed weight Wt[Npad][ldk] bf16:
// lane = N, lanes 0-15 hold K kb..kb+15, lanes 16-31 K kb+16..kb+31 -> one 32B load.
__device__ __forceinline__ v16bf load_b_t(const __bf16* __restrict__ Wt, int ldk,
                                          int n0, int ks, int lane) {
  int n = n0 + (lane & 15);
  int kb = ks * 32 + ((lane < 16) ? 0 : 16);
  return *(const v16bf*)(Wt + (size_t)n * ldk + kb);
}

// ---------------- kernel 0: pack weights to padded bf16 [N][K], zero ctx pad ----------------
__global__ __launch_bounds__(256) void k_pack(
    const float* __restrict__ Wq, const float* __restrict__ Wk,
    const float* __restrict__ Wv, const float* __restrict__ Wo,
    const float* __restrict__ W1, const float* __restrict__ W2,
    __bf16* __restrict__ WqT, __bf16* __restrict__ WkT,
    __bf16* __restrict__ WvT, __bf16* __restrict__ WoT,
    __bf16* __restrict__ W1T, __bf16* __restrict__ W2T,
    __bf16* __restrict__ ctxp) {
  int tid = blockIdx.x * blockDim.x + threadIdx.x;
  int stride = gridDim.x * blockDim.x;
  for (int i = tid; i < 48 * KP; i += stride) {
    int n = i / KP, k = i % KP;
    float q = 0.f, kk = 0.f, v = 0.f, o = 0.f;
    if (n < HID_ && k < HID_) {
      q = Wq[k * HID_ + n]; kk = Wk[k * HID_ + n];
      v = Wv[k * HID_ + n]; o = Wo[k * HID_ + n];
    }
    WqT[i] = f2bf(q); WkT[i] = f2bf(kk); WvT[i] = f2bf(v); WoT[i] = f2bf(o);
  }
  for (int i = tid; i < 32 * KP; i += stride) {
    int n = i / KP, k = i % KP;
    W1T[i] = f2bf((n < INTER_ && k < HID_) ? W1[k * INTER_ + n] : 0.f);
  }
  for (int i = tid; i < 48 * 32; i += stride) {
    int n = i / 32, k = i % 32;
    W2T[i] = f2bf((n < HID_ && k < INTER_) ? W2[k * HID_ + n] : 0.f);
  }
  for (int i = tid; i < B_ * S_ * KP; i += stride) ctxp[i] = f2bf(0.f);
}

// ---------------- kernel 1: LN1 + QKV projection ----------------
__global__ __launch_bounds__(96) void k_ln_qkv(
    const float* __restrict__ x, const float* __restrict__ g1,
    const float* __restrict__ bln1, const float* __restrict__ bq,
    const float* __restrict__ bk, const float* __restrict__ bv,
    const __bf16* __restrict__ WqT, const __bf16* __restrict__ WkT,
    const __bf16* __restrict__ WvT, __bf16* __restrict__ qpad,
    __bf16* __restrict__ kpad, __bf16* __restrict__ vpad) {
  __shared__ float xrow[16][HID_];
  __shared__ float mu[16], rs[16];
  __shared__ __bf16 Abuf[16][KP];
  int tid = threadIdx.x, lane = tid & 31, wave = tid >> 5;
  int m0 = blockIdx.x * 16;

  __builtin_prefetch(x + (size_t)(m0 + 16) * HID_, 0, 1);

  if (wave == 0) {
    tdm_load_2d((unsigned)(size_t)&xrow[0][0], x + (size_t)m0 * HID_,
                /*f32*/2, HID_, 16, HID_);
    __builtin_amdgcn_s_wait_tensorcnt(0);
  }
  // zero padded head-dims (ws is poisoned by harness)
  for (int i = tid; i < 16 * HEADS_ * (DHP - DH_); i += 96) {
    int r = i / (HEADS_ * (DHP - DH_));
    int rem = i % (HEADS_ * (DHP - DH_));
    int hh = rem / (DHP - DH_), d = DH_ + rem % (DHP - DH_);
    int m = m0 + r, bb = m / S_, s = m % S_;
    size_t idx = (((size_t)bb * HEADS_ + hh) * S_ + s) * DHP + d;
    qpad[idx] = f2bf(0.0f); kpad[idx] = f2bf(0.0f); vpad[idx] = f2bf(0.0f);
  }
  __syncthreads();
  if (tid < 16) {
    float s = 0.0f;
    for (int c = 0; c < HID_; ++c) s += xrow[tid][c];
    float m = s / HID_;
    float v = 0.0f;
    for (int c = 0; c < HID_; ++c) { float d = xrow[tid][c] - m; v += d * d; }
    mu[tid] = m;
    rs[tid] = rsqrtf(v / HID_ + EPS_);
  }
  __syncthreads();
  for (int i = tid; i < 16 * KP; i += 96) {
    int r = i / KP, c = i % KP;
    float hv = (c < HID_) ? (xrow[r][c] - mu[r]) * rs[r] * g1[c] + bln1[c] : 0.0f;
    Abuf[r][c] = f2bf(hv);
  }
  __syncthreads();

  const __bf16* Wt  = (wave == 0) ? WqT : (wave == 1) ? WkT : WvT;
  const float* bias = (wave == 0) ? bq : (wave == 1) ? bk : bv;
  __bf16* outp      = (wave == 0) ? qpad : (wave == 1) ? kpad : vpad;

  v16bf a0 = load_a(&Abuf[0][0], KP, 0, lane);
  v16bf a1 = load_a(&Abuf[0][0], KP, 1, lane);
#pragma unroll
  for (int nt = 0; nt < 3; ++nt) {
    v16bf w0 = load_b_t(Wt, KP, nt * 16, 0, lane);
    v16bf w1 = load_b_t(Wt, KP, nt * 16, 1, lane);
    v8f c = {};
    c = wmma_bf16(a0, w0, c);
    c = wmma_bf16(a1, w1, c);
    int n = nt * 16 + (lane & 15);
    if (n < HID_) {
      int hh = n / DH_, d = n % DH_;
      float bn = bias[n];
#pragma unroll
      for (int r = 0; r < 8; ++r) {
        int m = m0 + r + ((lane < 16) ? 0 : 8);
        int bb = m / S_, s = m % S_;
        outp[(((size_t)bb * HEADS_ + hh) * S_ + s) * DHP + d] = f2bf(c[r] + bn);
      }
    }
  }
}

// ---------------- kernel 2: attention (scores -> softmax -> ctx) ----------------
__global__ __launch_bounds__(128) void k_attn(const __bf16* __restrict__ qpad,
                                              const __bf16* __restrict__ kpad,
                                              const __bf16* __restrict__ vpad,
                                              __bf16* __restrict__ ctxp) {
  __shared__ __bf16 Klds[S_][DHP];     // 16 KB, TDM-filled
  __shared__ __bf16 Vlds[S_][DHP];     // 16 KB, TDM-filled
  __shared__ __bf16 Vt[DHP][S_];       // 16 KB, transposed V (dim-major)
  __shared__ float  sc[16][S_];        // 32 KB
  __shared__ __bf16 P[16][S_];         // 16 KB
  __shared__ float  part[4][16][16];   // 4 KB
  int blk = blockIdx.x;
  int qb = blk & 31, h = (blk >> 5) & 3, b = blk >> 7;
  int tid = threadIdx.x, lane = tid & 31, wave = tid >> 5;
  int s0 = qb * 16;
  const __bf16* qbase = qpad + ((size_t)b * HEADS_ + h) * S_ * DHP;
  const __bf16* kbase = kpad + ((size_t)b * HEADS_ + h) * S_ * DHP;
  const __bf16* vbase = vpad + ((size_t)b * HEADS_ + h) * S_ * DHP;

  if (wave == 0) {
    tdm_load_2d((unsigned)(size_t)&Klds[0][0], kbase, /*bf16*/1, DHP, S_, DHP);
    tdm_load_2d((unsigned)(size_t)&Vlds[0][0], vbase, /*bf16*/1, DHP, S_, DHP);
    __builtin_amdgcn_s_wait_tensorcnt(0);
  }

  // A fragment: Q rows (lanes 0-15 one 32B global load of K 0..15; lanes 16-31 zero, K>=16 is pad)
  v16bf aq;
  if (lane < 16) aq = *(const v16bf*)(qbase + (size_t)(s0 + lane) * DHP);
  else           aq = zero_v16bf();
  __syncthreads();

  // transpose V into dim-major layout for contiguous ctx B-fragments
  for (int i = tid; i < DHP * S_; i += 128) {
    int d = i >> 9, kpos = i & (S_ - 1);
    Vt[d][kpos] = Vlds[kpos][d];
  }

  const float scale = 0.316227766016838f;  // 1/sqrt(10)
#pragma unroll
  for (int tt = 0; tt < 8; ++tt) {
    int t = tt * 4 + wave;
    v16bf bk_;
    if (lane < 16) bk_ = *(const v16bf*)&Klds[t * 16 + lane][0];
    else           bk_ = zero_v16bf();
    v8f c = {};
    c = wmma_bf16(aq, bk_, c);
    int n = lane & 15;
#pragma unroll
    for (int r = 0; r < 8; ++r)
      sc[r + ((lane < 16) ? 0 : 8)][t * 16 + n] = c[r] * scale;
  }
  __syncthreads();
  // softmax: wave w handles rows w, w+4, w+8, w+12
  for (int rr = 0; rr < 4; ++rr) {
    int row = wave + rr * 4;
    float mx = -3.4e38f;
    for (int j = lane; j < S_; j += 32) mx = fmaxf(mx, sc[row][j]);
#pragma unroll
    for (int o = 16; o; o >>= 1) mx = fmaxf(mx, __shfl_xor(mx, o, 32));
    float sum = 0.0f;
    for (int j = lane; j < S_; j += 32) {
      float e = __expf(sc[row][j] - mx);
      sc[row][j] = e;
      sum += e;
    }
#pragma unroll
    for (int o = 16; o; o >>= 1) sum += __shfl_xor(sum, o, 32);
    float inv = 1.0f / sum;
    for (int j = lane; j < S_; j += 32) P[row][j] = f2bf(sc[row][j] * inv);
  }
  __syncthreads();
  // ctx = P @ V : K=512 split across 4 waves (4 k-steps each), LDS reduce
  v8f c = {};
  int dim = lane & 15;
#pragma unroll
  for (int u = 0; u < 4; ++u) {
    int kk = u * 4 + wave;
    v16bf ap = load_a(&P[0][0], S_, kk, lane);
    int kb = kk * 32 + ((lane < 16) ? 0 : 16);
    v16bf bv_ = *(const v16bf*)&Vt[dim][kb];
    c = wmma_bf16(ap, bv_, c);
  }
#pragma unroll
  for (int r = 0; r < 8; ++r) part[wave][r + ((lane < 16) ? 0 : 8)][dim] = c[r];
  __syncthreads();
  for (int i = tid; i < 16 * DH_; i += 128) {
    int r = i / DH_, d = i % DH_;
    float v = part[0][r][d] + part[1][r][d] + part[2][r][d] + part[3][r][d];
    ctxp[((size_t)b * S_ + s0 + r) * KP + h * DH_ + d] = f2bf(v);
  }
}

// ---------------- kernel 3: out-proj + residual + LN2 + FFN + residual ----------------
__global__ __launch_bounds__(96) void k_out_ffn(
    const __bf16* __restrict__ ctxp, const float* __restrict__ x,
    const __bf16* __restrict__ WoT, const float* __restrict__ bo,
    const float* __restrict__ g2, const float* __restrict__ bln2,
    const __bf16* __restrict__ W1T, const float* __restrict__ bb1,
    const __bf16* __restrict__ W2T, const float* __restrict__ bb2,
    float* __restrict__ out) {
  __shared__ __bf16 Actx[16][KP];
  __shared__ float  ao[16][HID_];
  __shared__ float  mu[16], rs[16];
  __shared__ __bf16 Ah2[16][KP];
  __shared__ __bf16 Ai[16][32];
  int tid = threadIdx.x, lane = tid & 31, wave = tid >> 5;
  int m0 = blockIdx.x * 16;

  if (wave == 0) {
    tdm_load_2d((unsigned)(size_t)&Actx[0][0], ctxp + (size_t)m0 * KP,
                /*bf16*/1, KP, 16, KP);
    __builtin_amdgcn_s_wait_tensorcnt(0);
  }
  __syncthreads();
  // attn_out = ctx @ Wo + bo + x
  {
    v16bf a0 = load_a(&Actx[0][0], KP, 0, lane);
    v16bf a1 = load_a(&Actx[0][0], KP, 1, lane);
    v16bf w0 = load_b_t(WoT, KP, wave * 16, 0, lane);
    v16bf w1 = load_b_t(WoT, KP, wave * 16, 1, lane);
    v8f c = {};
    c = wmma_bf16(a0, w0, c);
    c = wmma_bf16(a1, w1, c);
    int n = wave * 16 + (lane & 15);
    if (n < HID_) {
      float bn = bo[n];
#pragma unroll
      for (int r = 0; r < 8; ++r) {
        int m = r + ((lane < 16) ? 0 : 8);
        ao[m][n] = c[r] + bn + x[(size_t)(m0 + m) * HID_ + n];
      }
    }
  }
  __syncthreads();
  if (tid < 16) {
    float s = 0.0f;
    for (int c = 0; c < HID_; ++c) s += ao[tid][c];
    float m = s / HID_;
    float v = 0.0f;
    for (int c = 0; c < HID_; ++c) { float d = ao[tid][c] - m; v += d * d; }
    mu[tid] = m;
    rs[tid] = rsqrtf(v / HID_ + EPS_);
  }
  __syncthreads();
  for (int i = tid; i < 16 * KP; i += 96) {
    int r = i / KP, c = i % KP;
    Ah2[r][c] = (c < HID_) ? f2bf((ao[r][c] - mu[r]) * rs[r] * g2[c] + bln2[c])
                           : f2bf(0.0f);
  }
  __syncthreads();
  // inter = gelu(h2 @ W1 + b1), N padded 20->32, waves 0..1
  if (wave < 2) {
    v16bf a0 = load_a(&Ah2[0][0], KP, 0, lane);
    v16bf a1 = load_a(&Ah2[0][0], KP, 1, lane);
    v16bf w0 = load_b_t(W1T, KP, wave * 16, 0, lane);
    v16bf w1 = load_b_t(W1T, KP, wave * 16, 1, lane);
    v8f c = {};
    c = wmma_bf16(a0, w0, c);
    c = wmma_bf16(a1, w1, c);
    int n = wave * 16 + (lane & 15);
#pragma unroll
    for (int r = 0; r < 8; ++r) {
      int m = r + ((lane < 16) ? 0 : 8);
      float v = 0.0f;
      if (n < INTER_) {
        float t = c[r] + bb1[n];
        v = 0.5f * t * (1.0f + erff(t * 0.707106781186547f));
      }
      Ai[m][n] = f2bf(v);
    }
  }
  __syncthreads();
  // out = inter @ W2 + b2 + attn_out  (K padded 20->32, single k-step)
  {
    v16bf a = load_a(&Ai[0][0], 32, 0, lane);
    v16bf w = load_b_t(W2T, 32, wave * 16, 0, lane);
    v8f c = {};
    c = wmma_bf16(a, w, c);
    int n = wave * 16 + (lane & 15);
    if (n < HID_) {
      float bn = bb2[n];
#pragma unroll
      for (int r = 0; r < 8; ++r) {
        int m = r + ((lane < 16) ? 0 : 8);
        out[(size_t)(m0 + m) * HID_ + n] = c[r] + bn + ao[m][n];
      }
    }
  }
}

extern "C" void kernel_launch(void* const* d_in, const int* in_sizes, int n_in,
                              void* d_out, int out_size, void* d_ws,
                              size_t ws_size, hipStream_t stream) {
  (void)in_sizes; (void)n_in; (void)out_size; (void)ws_size;
  const float* x    = (const float*)d_in[0];
  const float* g1   = (const float*)d_in[1];
  const float* bln1 = (const float*)d_in[2];
  const float* Wq   = (const float*)d_in[3];
  const float* bq   = (const float*)d_in[4];
  const float* Wk   = (const float*)d_in[5];
  const float* bk   = (const float*)d_in[6];
  const float* Wv   = (const float*)d_in[7];
  const float* bv   = (const float*)d_in[8];
  const float* Wo   = (const float*)d_in[9];
  const float* bo   = (const float*)d_in[10];
  const float* g2   = (const float*)d_in[11];
  const float* bln2 = (const float*)d_in[12];
  const float* W1   = (const float*)d_in[13];
  const float* bb1  = (const float*)d_in[14];
  const float* W2   = (const float*)d_in[15];
  const float* bb2  = (const float*)d_in[16];
  float* out = (float*)d_out;

  char* ws = (char*)d_ws;
  size_t off = 0;
  const size_t qkv_bytes = (size_t)B_ * HEADS_ * S_ * DHP * sizeof(__bf16); // 4 MB
  __bf16* qpad = (__bf16*)(ws + off); off += qkv_bytes;
  __bf16* kpad = (__bf16*)(ws + off); off += qkv_bytes;
  __bf16* vpad = (__bf16*)(ws + off); off += qkv_bytes;
  __bf16* ctxp = (__bf16*)(ws + off); off += (size_t)B_ * S_ * KP * 2;      // 4 MB
  __bf16* WqT  = (__bf16*)(ws + off); off += 48 * KP * 2;
  __bf16* WkT  = (__bf16*)(ws + off); off += 48 * KP * 2;
  __bf16* WvT  = (__bf16*)(ws + off); off += 48 * KP * 2;
  __bf16* WoT  = (__bf16*)(ws + off); off += 48 * KP * 2;
  __bf16* W1T  = (__bf16*)(ws + off); off += 32 * KP * 2;
  __bf16* W2T  = (__bf16*)(ws + off); off += 48 * 32 * 2;

  const int mblocks = (B_ * S_) / 16;  // 2048
  k_pack<<<64, 256, 0, stream>>>(Wq, Wk, Wv, Wo, W1, W2, WqT, WkT, WvT, WoT,
                                 W1T, W2T, ctxp);
  k_ln_qkv<<<mblocks, 96, 0, stream>>>(x, g1, bln1, bq, bk, bv, WqT, WkT, WvT,
                                       qpad, kpad, vpad);
  k_attn<<<B_ * HEADS_ * (S_ / 16), 128, 0, stream>>>(qpad, kpad, vpad, ctxp);
  k_out_ffn<<<mblocks, 96, 0, stream>>>(ctxp, x, WoT, bo, g2, bln2, W1T, bb1,
                                        W2T, bb2, out);
}
